// GPT2Block_35691178230208
// MI455X (gfx1250) — compile-verified, matrix operand order checked by reading, not run on hardware
//
#include <hip/hip_runtime.h>

// ---------------- problem constants (match reference) ----------------
constexpr int kB  = 2;
constexpr int kS  = 2048;
constexpr int kD  = 1024;
constexpr int kH  = 16;
constexpr int kHD = 64;
constexpr int kDF = 4096;
constexpr int kM  = kB * kS;        // 4096 rows

// ---------------- types ----------------
typedef __attribute__((ext_vector_type(16))) __bf16          v16bf;
typedef __attribute__((ext_vector_type(16))) unsigned short  v16u;
typedef __attribute__((ext_vector_type(8)))  float           v8f;
typedef __attribute__((ext_vector_type(4)))  unsigned int    v4u;
typedef __attribute__((ext_vector_type(8)))  int             v8i;
typedef __attribute__((ext_vector_type(4)))  int             v4i;

#if defined(__has_builtin)
#  if __has_builtin(__builtin_amdgcn_tensor_load_to_lds)
#    define HAVE_TDM 1
#  endif
#endif
#ifndef HAVE_TDM
#  define HAVE_TDM 0
#endif

__device__ __forceinline__ unsigned short f2bf(float f) {
    unsigned int u = __builtin_bit_cast(unsigned int, f);
    unsigned int r = u + 0x7FFFu + ((u >> 16) & 1u);   // round-to-nearest-even
    return (unsigned short)(r >> 16);
}

// ---------------- WMMA fragment loaders (wave32, 16x16x32 bf16) -------------
// A-fragment (16x32, row-major source, K contiguous):
//   lane L: row = L%16 ; VGPR i holds K pair: K = (i/4)*16 + (L/16)*8 + (i%4)*2
__device__ __forceinline__ v16bf load_frag_rowmajor(const unsigned short* p, int stride) {
    int lane = (int)(threadIdx.x & 31);
    int rr = lane & 15, hh = lane >> 4;
    const unsigned short* q = p + (size_t)rr * stride;
    union { unsigned int wi[8]; v16bf bb; } cv;
#pragma unroll
    for (int i = 0; i < 8; ++i) {
        int k = ((i >> 2) << 4) + (hh << 3) + ((i & 3) << 1);
        cv.wi[i] = *(const unsigned int*)(q + k);       // packed bf16 pair
    }
    return cv.bb;
}

// B-fragment (32x16, K x N source with row stride = `stride`, N contiguous):
__device__ __forceinline__ v16bf load_bfrag_strided(const unsigned short* p, int stride) {
    int lane = (int)(threadIdx.x & 31);
    int nn = lane & 15, hh = lane >> 4;
    v16u u;
#pragma unroll
    for (int i = 0; i < 8; ++i) {
        int k = ((i >> 2) << 4) + (hh << 3) + ((i & 3) << 1);
        u[2 * i]     = p[(size_t)k * stride + nn];
        u[2 * i + 1] = p[(size_t)(k + 1) * stride + nn];
    }
    union { v16u uu; v16bf bb; } cv; cv.uu = u;
    return cv.bb;
}

__device__ __forceinline__ v8f wmma_bf16(v16bf a, v16bf b, v8f c) {
    return __builtin_amdgcn_wmma_f32_16x16x32_bf16(false, a, false, b, (short)0, c,
                                                   false, false);
}

#if HAVE_TDM
// ---------------- Tensor Data Mover: 2D bf16 tile, global -> LDS ------------
// D# per CDNA5 ISA ch.8. data_size=1 (2B). Padding: after every 32 dwords
// (one 64-elem row = 128B) insert 4 dwords (16B) -> LDS row stride 144B,
// which spreads the 16-lane fragment reads across banks.
__device__ __forceinline__ void tdm_load_2d_bf16(
        const unsigned short* gptr, unsigned lds_off,
        unsigned tensor_d0, unsigned tensor_d1,
        unsigned long long stride0, unsigned tile_d0, unsigned tile_d1) {
    unsigned long long ga = (unsigned long long)(uintptr_t)gptr;
    v4u g0;
    g0[0] = 1u;                                            // count=1, user descriptor
    g0[1] = lds_off;                                       // LDS byte address
    g0[2] = (unsigned)(ga & 0xffffffffu);                  // global_addr[31:0]
    g0[3] = (unsigned)((ga >> 32) & 0x01ffffffu) | (2u << 30);  // addr[56:32] | type=2
    v8i g1;
    g1[0] = (int)((1u << 16)        // data_size = 1 -> 2 bytes
                | (1u << 20)        // pad_enable
                | (4u << 22)        // pad_interval: 32 dwords
                | (3u << 25));      // pad_amount:   4 dwords (16B)
    g1[1] = (int)((tensor_d0 & 0xffffu) << 16);
    g1[2] = (int)((tensor_d0 >> 16) | ((tensor_d1 & 0xffffu) << 16));
    g1[3] = (int)((tensor_d1 >> 16) | (tile_d0 << 16));
    g1[4] = (int)tile_d1;                                  // tile_dim2 = 0
    g1[5] = (int)(unsigned)(stride0 & 0xffffffffu);
    g1[6] = (int)(unsigned)((stride0 >> 32) & 0xffffu);    // tensor_dim1_stride = 0
    g1[7] = 0;
    v4i z4 = {0, 0, 0, 0};
    v8i z8 = {0, 0, 0, 0, 0, 0, 0, 0};
    __builtin_amdgcn_tensor_load_to_lds(g0, g1, z4, z4, z8, 0);  // 6-arg (clang-23)
}
#endif

// ---------------- weight conversion: f32 [K,N] -> bf16 [N,K] ----------------
__global__ __launch_bounds__(256) void cvt_tr_kernel(const float* __restrict__ in,
                                                     unsigned short* __restrict__ out,
                                                     int K, int N) {
    int idx = blockIdx.x * 256 + threadIdx.x;   // idx < K*N (exact multiple)
    int k = idx / N, n = idx % N;
    out[(size_t)n * K + k] = f2bf(in[idx]);
}

// ---------------- LayerNorm fused with bf16 cast ----------------
__global__ __launch_bounds__(256) void ln_bf16_kernel(const float* __restrict__ x,
                                                      const float* __restrict__ w,
                                                      const float* __restrict__ b,
                                                      unsigned short* __restrict__ out) {
    int row = blockIdx.x;
    const float* xr = x + (size_t)row * kD;
    float s = 0.f, s2 = 0.f;
    for (int i = threadIdx.x; i < kD; i += 256) {
        float v = xr[i]; s += v; s2 += v * v;
    }
#pragma unroll
    for (int m = 16; m >= 1; m >>= 1) {
        s  += __shfl_xor(s,  m, 32);
        s2 += __shfl_xor(s2, m, 32);
    }
    __shared__ float red[2][8];
    int wv = threadIdx.x >> 5, lane = threadIdx.x & 31;
    if (lane == 0) { red[0][wv] = s; red[1][wv] = s2; }
    __syncthreads();
    float ts = 0.f, ts2 = 0.f;
#pragma unroll
    for (int i = 0; i < 8; ++i) { ts += red[0][i]; ts2 += red[1][i]; }
    float mu  = ts / (float)kD;
    float var = ts2 / (float)kD - mu * mu;
    float inv = rsqrtf(var + 1e-5f);
    unsigned short* orow = out + (size_t)row * kD;
    for (int i = threadIdx.x; i < kD; i += 256) {
        float v = (xr[i] - mu) * inv * w[i] + b[i];
        orow[i] = f2bf(v);
    }
}

// ---------------- WMMA GEMM: TDM double-buffered LDS staging ----------------
// C[M,N] = A[M,K](bf16) @ B (bf16, stored transposed N x K) + bias
//          (+ resid f32) (+ tanh-GELU); writes f32 (Cf) and/or bf16 (Cb).
// Block tile 64x128, 8 waves (4 row x 2 col), k-stage = 64.
// Requires M%64==0, N%128==0, K%64==0 (true for all calls here).
constexpr int KST   = 64;                 // k elements per stage
constexpr int LROW  = KST + 8;            // LDS row stride in elems (144B, padded)

__global__ __launch_bounds__(256) void gemm_bf16_kernel(
        const unsigned short* __restrict__ A,
        const unsigned short* __restrict__ Bt,
        const float* __restrict__ bias,
        const float* __restrict__ resid,
        float* __restrict__ Cf,
        unsigned short* __restrict__ Cb,
        int M, int N, int K, int gelu) {
    int w  = (int)(threadIdx.x >> 5);
    int wm = w >> 1, wn = w & 1;
    int row0 = blockIdx.y * 64;
    int col0 = blockIdx.x * 128;

    v8f acc[4] = {};

#if HAVE_TDM
    __shared__ __align__(16) unsigned short As[2][64  * LROW];   // 2 x 9  KB
    __shared__ __align__(16) unsigned short Bs[2][128 * LROW];   // 2 x 18 KB
    const int nst = K / KST;
    unsigned aoff[2] = { (unsigned)(uintptr_t)&As[0][0], (unsigned)(uintptr_t)&As[1][0] };
    unsigned boff[2] = { (unsigned)(uintptr_t)&Bs[0][0], (unsigned)(uintptr_t)&Bs[1][0] };

    if (w == 0) {
        // prime both buffers
        tdm_load_2d_bf16(A  + (size_t)row0 * K,            aoff[0], K, M, K, KST, 64);
        tdm_load_2d_bf16(Bt + (size_t)col0 * K,            boff[0], K, N, K, KST, 128);
        if (nst > 1) {
            tdm_load_2d_bf16(A  + (size_t)row0 * K + KST,  aoff[1], K, M, K, KST, 64);
            tdm_load_2d_bf16(Bt + (size_t)col0 * K + KST,  boff[1], K, N, K, KST, 128);
        }
    }
    for (int st = 0; st < nst; ++st) {
        int buf = st & 1;
        if (w == 0) {
            if (st + 1 < nst) __builtin_amdgcn_s_wait_tensorcnt(2);  // current pair done
            else              __builtin_amdgcn_s_wait_tensorcnt(0);  // drain on last stage
        }
        __syncthreads();                       // stage `st` resident in LDS
#pragma unroll
        for (int kk = 0; kk < KST; kk += 32) {
            v16bf af = load_frag_rowmajor(&As[buf][(size_t)(wm * 16) * LROW + kk], LROW);
#pragma unroll
            for (int t = 0; t < 4; ++t) {
                v16bf bf = load_frag_rowmajor(
                        &Bs[buf][(size_t)(wn * 64 + t * 16) * LROW + kk], LROW);
                acc[t] = wmma_bf16(af, bf, acc[t]);
            }
        }
        __syncthreads();                       // done reading `buf`
        if (w == 0 && st + 2 < nst) {          // refill `buf` with stage st+2
            size_t kofs = (size_t)(st + 2) * KST;
            tdm_load_2d_bf16(A  + (size_t)row0 * K + kofs, aoff[buf], K, M, K, KST, 64);
            tdm_load_2d_bf16(Bt + (size_t)col0 * K + kofs, boff[buf], K, N, K, KST, 128);
        }
    }
#else
    // fallback: direct-from-global fragment streaming
    int ar0 = row0 + wm * 16, bc0 = col0 + wn * 64;
    for (int k0 = 0; k0 < K; k0 += 32) {
        v16bf af = load_frag_rowmajor(A + (size_t)ar0 * K + k0, K);
#pragma unroll
        for (int t = 0; t < 4; ++t) {
            v16bf bf = load_frag_rowmajor(Bt + (size_t)(bc0 + t * 16) * K + k0, K);
            acc[t] = wmma_bf16(af, bf, acc[t]);
        }
    }
#endif

    int lane = (int)(threadIdx.x & 31);
    int n = lane & 15, hi = lane >> 4;
#pragma unroll
    for (int t = 0; t < 4; ++t) {
        int col = col0 + wn * 64 + t * 16 + n;
        float bb = bias[col];
#pragma unroll
        for (int r = 0; r < 8; ++r) {
            int row = row0 + wm * 16 + r + hi * 8;
            float v = acc[t][r] + bb;
            if (resid) v += resid[(size_t)row * N + col];
            if (gelu) {
                float xx = v;
                v = 0.5f * xx * (1.0f + tanhf(0.7978845608f * (xx + 0.044715f * xx * xx * xx)));
            }
            if (Cf) Cf[(size_t)row * N + col] = v;
            if (Cb) Cb[(size_t)row * N + col] = f2bf(v);
        }
    }
}

// ---------------- flash attention (causal), one 16-row Q tile per wave ------
__global__ __launch_bounds__(256) void attn_flash_kernel(
        const unsigned short* __restrict__ Q,
        const unsigned short* __restrict__ Km,
        const unsigned short* __restrict__ Vm,
        unsigned short* __restrict__ O) {
    __shared__ __align__(16) unsigned short pshm[8][16 * 32];   // per-wave P tile (bf16)

    int w    = (int)(threadIdx.x >> 5);
    int lane = (int)(threadIdx.x & 31);
    int n = lane & 15, hi = lane >> 4;

    int bh = blockIdx.x;                 // 0 .. B*H-1
    int b  = bh >> 4, h = bh & (kH - 1);
    int qt = blockIdx.y * 8 + w;         // q tile index, 0 .. S/16-1
    int q0 = qt * 16;

    const unsigned short* qp = Q + ((size_t)b * kS + q0) * kD + h * kHD;
    const unsigned short* kp = Km + (size_t)b * kS * kD + h * kHD;
    const unsigned short* vp = Vm + (size_t)b * kS * kD + h * kHD;

    v16bf qf0 = load_frag_rowmajor(qp, kD);
    v16bf qf1 = load_frag_rowmajor(qp + 32, kD);

    v8f o[4] = {};
    float m[8], l[8];
#pragma unroll
    for (int r = 0; r < 8; ++r) { m[r] = -3.0e38f; l[r] = 0.f; }

    const float sc = 0.125f;             // 1/sqrt(HD)
    int nb = (q0 + 16 + 31) / 32;        // causal kv blocks of 32
    unsigned short* pw = &pshm[w][0];

    for (int j = 0; j < nb; ++j) {
        int j0 = j * 32;

        v8f sfr[2];
#pragma unroll
        for (int t = 0; t < 2; ++t) {
            const unsigned short* kt = kp + (size_t)(j0 + t * 16) * kD;
            v16bf kb0 = load_frag_rowmajor(kt, kD);        // B[kk][n] = K[n_row][kk]
            v16bf kb1 = load_frag_rowmajor(kt + 32, kD);
            v8f c = {};
            c = wmma_bf16(qf0, kb0, c);
            c = wmma_bf16(qf1, kb1, c);
            sfr[t] = c;
        }

        bool full = (j0 + 31) <= q0;
        float sv[2][8], rmax[8];
#pragma unroll
        for (int r = 0; r < 8; ++r) rmax[r] = -3.0e38f;
#pragma unroll
        for (int t = 0; t < 2; ++t)
#pragma unroll
            for (int r = 0; r < 8; ++r) {
                float v = sfr[t][r] * sc;
                if (!full) {
                    int col = j0 + t * 16 + n;
                    int row = q0 + r + hi * 8;
                    if (col > row) v = -3.0e38f;
                }
                sv[t][r] = v;
                rmax[r] = fmaxf(rmax[r], v);
            }
#pragma unroll
        for (int r = 0; r < 8; ++r)
#pragma unroll
            for (int msk = 1; msk < 16; msk <<= 1)
                rmax[r] = fmaxf(rmax[r], __shfl_xor(rmax[r], msk, 32));

        float alpha[8];
#pragma unroll
        for (int r = 0; r < 8; ++r) {
            float mn = fmaxf(m[r], rmax[r]);
            alpha[r] = __expf(m[r] - mn);
            m[r] = mn;
        }
        float rsum[8];
#pragma unroll
        for (int r = 0; r < 8; ++r) rsum[r] = 0.f;
#pragma unroll
        for (int t = 0; t < 2; ++t)
#pragma unroll
            for (int r = 0; r < 8; ++r) {
                float p = __expf(sv[t][r] - m[r]);
                rsum[r] += p;
                pw[(size_t)(r + hi * 8) * 32 + t * 16 + n] = f2bf(p);
            }
#pragma unroll
        for (int r = 0; r < 8; ++r) {
#pragma unroll
            for (int msk = 1; msk < 16; msk <<= 1)
                rsum[r] += __shfl_xor(rsum[r], msk, 32);
            l[r] = l[r] * alpha[r] + rsum[r];
        }
#pragma unroll
        for (int t = 0; t < 4; ++t)
#pragma unroll
            for (int r = 0; r < 8; ++r)
                o[t][r] = o[t][r] * alpha[r];

        asm volatile("s_wait_dscnt 0" ::: "memory");   // LDS RAW, per-wave

        v16bf pa = load_frag_rowmajor(pw, 32);
#pragma unroll
        for (int cf = 0; cf < 4; ++cf) {
            v16bf vb = load_bfrag_strided(vp + (size_t)j0 * kD + cf * 16, kD);
            o[cf] = wmma_bf16(pa, vb, o[cf]);
        }
    }

    unsigned short* op = O + ((size_t)b * kS + q0) * kD + h * kHD;
    float inv[8];
#pragma unroll
    for (int r = 0; r < 8; ++r) inv[r] = 1.0f / l[r];
#pragma unroll
    for (int t = 0; t < 4; ++t)
#pragma unroll
        for (int r = 0; r < 8; ++r)
            op[(size_t)(r + hi * 8) * kD + t * 16 + n] = f2bf(o[t][r] * inv[r]);
}

// ---------------- host side ----------------
extern "C" void kernel_launch(void* const* d_in, const int* in_sizes, int n_in,
                              void* d_out, int out_size, void* d_ws, size_t ws_size,
                              hipStream_t stream) {
    (void)in_sizes; (void)n_in; (void)out_size; (void)ws_size;

    const float* x    = (const float*)d_in[0];
    const float* ln1w = (const float*)d_in[1];
    const float* ln1b = (const float*)d_in[2];
    const float* Wq   = (const float*)d_in[3];
    const float* bq   = (const float*)d_in[4];
    const float* Wk   = (const float*)d_in[5];
    const float* bk   = (const float*)d_in[6];
    const float* Wv   = (const float*)d_in[7];
    const float* bv   = (const float*)d_in[8];
    const float* Wo   = (const float*)d_in[9];
    const float* bo   = (const float*)d_in[10];
    const float* ln2w = (const float*)d_in[11];
    const float* ln2b = (const float*)d_in[12];
    const float* Wfc  = (const float*)d_in[13];
    const float* bfc  = (const float*)d_in[14];
    const float* Wp   = (const float*)d_in[15];
    const float* bp   = (const float*)d_in[16];

    char* ws = (char*)d_ws;
    size_t off = 0;
    auto take = [&](size_t bytes) -> char* {
        char* p = ws + off;
        off += (bytes + 255) & ~(size_t)255;
        return p;
    };
    unsigned short* h1    = (unsigned short*)take((size_t)kM * kD * 2);
    unsigned short* qb    = (unsigned short*)take((size_t)kM * kD * 2);
    unsigned short* kb    = (unsigned short*)take((size_t)kM * kD * 2);
    unsigned short* vb    = (unsigned short*)take((size_t)kM * kD * 2);
    unsigned short* attnb = (unsigned short*)take((size_t)kM * kD * 2);
    float*          xres  = (float*)take((size_t)kM * kD * 4);
    unsigned short* h2    = (unsigned short*)take((size_t)kM * kD * 2);
    unsigned short* gb    = (unsigned short*)take((size_t)kM * kDF * 2);
    unsigned short* wqb   = (unsigned short*)take((size_t)kD * kD * 2);
    unsigned short* wkb   = (unsigned short*)take((size_t)kD * kD * 2);
    unsigned short* wvb   = (unsigned short*)take((size_t)kD * kD * 2);
    unsigned short* wob   = (unsigned short*)take((size_t)kD * kD * 2);
    unsigned short* wfcb  = (unsigned short*)take((size_t)kD * kDF * 2);
    unsigned short* wpb   = (unsigned short*)take((size_t)kDF * kD * 2);

    // 1) weight conversion (f32 [K,N] -> bf16 [N,K])
    cvt_tr_kernel<<<dim3(kD * kD / 256), 256, 0, stream>>>(Wq,  wqb,  kD,  kD);
    cvt_tr_kernel<<<dim3(kD * kD / 256), 256, 0, stream>>>(Wk,  wkb,  kD,  kD);
    cvt_tr_kernel<<<dim3(kD * kD / 256), 256, 0, stream>>>(Wv,  wvb,  kD,  kD);
    cvt_tr_kernel<<<dim3(kD * kD / 256), 256, 0, stream>>>(Wo,  wob,  kD,  kD);
    cvt_tr_kernel<<<dim3(kD * kDF / 256), 256, 0, stream>>>(Wfc, wfcb, kD,  kDF);
    cvt_tr_kernel<<<dim3(kDF * kD / 256), 256, 0, stream>>>(Wp,  wpb,  kDF, kD);

    // 2) LN1 -> bf16
    ln_bf16_kernel<<<dim3(kM), 256, 0, stream>>>(x, ln1w, ln1b, h1);

    // 3) QKV projections (WMMA + TDM staging)
    dim3 gD(kD / 128, kM / 64), gDF(kDF / 128, kM / 64);
    gemm_bf16_kernel<<<gD, 256, 0, stream>>>(h1, wqb, bq, nullptr, nullptr, qb, kM, kD, kD, 0);
    gemm_bf16_kernel<<<gD, 256, 0, stream>>>(h1, wkb, bk, nullptr, nullptr, kb, kM, kD, kD, 0);
    gemm_bf16_kernel<<<gD, 256, 0, stream>>>(h1, wvb, bv, nullptr, nullptr, vb, kM, kD, kD, 0);

    // 4) causal flash attention (WMMA, online softmax)
    attn_flash_kernel<<<dim3(kB * kH, kS / 128), 256, 0, stream>>>(qb, kb, vb, attnb);

    // 5) output projection + residual -> x_res (f32)
    gemm_bf16_kernel<<<gD, 256, 0, stream>>>(attnb, wob, bo, x, xres, nullptr, kM, kD, kD, 0);

    // 6) LN2 -> bf16
    ln_bf16_kernel<<<dim3(kM), 256, 0, stream>>>(xres, ln2w, ln2b, h2);

    // 7) FC + GELU -> bf16
    gemm_bf16_kernel<<<gDF, 256, 0, stream>>>(h2, wfcb, bfc, nullptr, nullptr, gb, kM, kDF, kD, 1);

    // 8) projection + residual -> d_out (f32)
    gemm_bf16_kernel<<<gD, 256, 0, stream>>>(gb, wpb, bp, xres, (float*)d_out, nullptr,
                                             kM, kD, kDF, 0);
}
